// EdgeConv_56573309223806
// MI455X (gfx1250) — compile-verified, matrix operand order checked by reading, not run on hardware
//
#include <hip/hip_runtime.h>
#include <stdint.h>

// ---------- CDNA5 vector types ----------
typedef __attribute__((ext_vector_type(2)))  float  v2f;
typedef __attribute__((ext_vector_type(8)))  float  v8f;
typedef __attribute__((ext_vector_type(8)))  __bf16 v8bf;
typedef __attribute__((ext_vector_type(16))) __bf16 v16bf;

#define B_   8
#define C_   64
#define N_   2048
#define K_   20
#define R_   (B_*N_*K_)          /* 327680 edge rows */
#define NEG_INF (-3.402823466e38f)

// Async global->LDS DMA (CDNA5 ASYNCcnt path). lds_addr = LDS byte address
// (low 32 bits of the flat address of a __shared__ pointer), gaddr = 64-bit VA.
__device__ __forceinline__ void async_g2l_b128(unsigned lds_addr, const void* gaddr) {
  unsigned long long ga = (unsigned long long)(uintptr_t)gaddr;
  asm volatile("global_load_async_to_lds_b128 %0, %1, off"
               :: "v"(lds_addr), "v"(ga) : "memory");
}
__device__ __forceinline__ void wait_async0() {
  asm volatile("s_wait_asynccnt 0" ::: "memory");
}

// ============================================================
// 0) zero BN stat accumulators (ws is poisoned by harness)
// stats layout (floats): [0:64) sum1 [64:128) sq1 [128:192) scale1
// [192:256) shift1 [256:320) sum2 [320:384) sq2 [384:448) scale2 [448:512) shift2
// ============================================================
__global__ void k_init_stats(float* __restrict__ stats) {
  int t = threadIdx.x;
  if (t < 512) stats[t] = 0.f;
}

// ============================================================
// 1) pack W1 (64x128) / W2 (64x64) into the per-lane bf16 B-operand
// layout of V_WMMA_*_16X16X32_BF16: element e of lane L holds
// K = kt*32 + (e<8 ? bs+e : bs+e+8), bs = (L>=16)?8:0 ; N = ot*16 + L%16
// ============================================================
__global__ void k_pack_weights(const float* __restrict__ W1, const float* __restrict__ W2,
                               __bf16* __restrict__ W1B, __bf16* __restrict__ W2B) {
  int u = blockIdx.x * 256 + threadIdx.x;
  if (u < 8192) {                                   // W1: kt 0..3, ot 0..3
    int e = u & 15, lane = (u >> 4) & 31, ot = (u >> 9) & 3, kt = u >> 11;
    int bs = (lane & 16) ? 8 : 0;
    int c = kt * 32 + ((e < 8) ? bs + e : bs + e + 8);
    int o = ot * 16 + (lane & 15);
    W1B[u] = (__bf16)W1[o * 128 + c];
  } else if (u < 12288) {                           // W2: kt 0..1, ot 0..3
    int u2 = u - 8192;
    int e = u2 & 15, lane = (u2 >> 4) & 31, ot = (u2 >> 9) & 3, kt = u2 >> 11;
    int bs = (lane & 16) ? 8 : 0;
    int c = kt * 32 + ((e < 8) ? bs + e : bs + e + 8);
    int o = ot * 16 + (lane & 15);
    W2B[u2] = (__bf16)W2[o * 64 + c];
  }
}

// ============================================================
// 2) squared norms  xx[b][n] = sum_c x[b][c][n]^2
// ============================================================
__global__ void k_norms(const float* __restrict__ x, float* __restrict__ xx) {
  int gid = blockIdx.x * 256 + threadIdx.x;         // 16384 threads
  int b = gid >> 11, n = gid & (N_ - 1);
  const float* p = x + (size_t)b * C_ * N_ + n;
  float s = 0.f;
  #pragma unroll
  for (int c = 0; c < C_; ++c) { float v = p[(size_t)c * N_]; s += v * v; }
  xx[gid] = s;
}

// ============================================================
// 3) pairwise pd = 2*x^T x - xx_i - xx_j  via V_WMMA_F32_16X16X4_F32
//    one wave per 16x16 tile, K-loop = 64/4 = 16; x (4 MB) is L2-resident
// A f32 16x4 layout: lane L: M=L%16 ; v0 K=(L<16?0:2), v1 K=(L<16?1:3)
// ============================================================
__global__ void k_dist_wmma(const float* __restrict__ x, const float* __restrict__ xx,
                            float* __restrict__ dist) {
  int wave = threadIdx.x >> 5, lane = threadIdx.x & 31;
  int tile = blockIdx.x * 8 + wave;                 // 131072 tiles
  int b  = tile >> 14;
  int it = (tile >> 7) & 127;
  int jt = tile & 127;
  int i0 = it * 16, j0 = jt * 16;
  int m  = lane & 15;
  int hi = (lane & 16) ? 1 : 0;
  const float* xb = x + (size_t)b * C_ * N_;
  v8f acc = {};
  #pragma unroll
  for (int kt = 0; kt < 16; ++kt) {
    int c0 = kt * 4 + 2 * hi;
    v2f a, bb;
    a.x  = xb[(size_t)c0 * N_ + i0 + m];            // A = x^T tile (coalesced in i)
    a.y  = xb[(size_t)(c0 + 1) * N_ + i0 + m];
    bb.x = xb[(size_t)c0 * N_ + j0 + m];            // B = x tile
    bb.y = xb[(size_t)(c0 + 1) * N_ + j0 + m];
    acc = __builtin_amdgcn_wmma_f32_16x16x4_f32(false, a, false, bb, (short)0,
                                                acc, false, false);
  }
  int j = j0 + m;
  float xj = xx[b * N_ + j];
  float* dbase = dist + (size_t)b * N_ * N_;
  #pragma unroll
  for (int v = 0; v < 8; ++v) {
    int i = i0 + v + 8 * hi;                        // C/D layout: M = v + (lane<16?0:8)
    float pd = 2.f * acc[v] - xx[b * N_ + i] - xj;
    dbase[(size_t)i * N_ + j] = pd;                 // coalesced in j
  }
}

// ============================================================
// 4) top-k=20 per row (thread per row, unrolled register insertion sort)
// ============================================================
__global__ void k_topk(const float* __restrict__ dist, int* __restrict__ idx) {
  int row = blockIdx.x * 256 + threadIdx.x;         // 16384 rows
  const float* p = dist + (size_t)row * N_;
  float tv[K_]; int ti[K_];
  #pragma unroll
  for (int s = 0; s < K_; ++s) { tv[s] = NEG_INF; ti[s] = 0; }
  for (int j = 0; j < N_; ++j) {
    float v = p[j];
    if (v > tv[K_ - 1]) {
      tv[K_ - 1] = v; ti[K_ - 1] = j;
      #pragma unroll
      for (int s = K_ - 1; s > 0; --s) {            // static-index compare/swap chain
        bool sw = tv[s] > tv[s - 1];
        float fa = sw ? tv[s - 1] : tv[s];
        float fb = sw ? tv[s]     : tv[s - 1];
        int   ia = sw ? ti[s - 1] : ti[s];
        int   ib = sw ? ti[s]     : ti[s - 1];
        tv[s] = fa; tv[s - 1] = fb; ti[s] = ia; ti[s - 1] = ib;
      }
    }
  }
  int* op = idx + (size_t)row * K_;
  #pragma unroll
  for (int s = 0; s < K_; ++s) op[s] = ti[s];
}

// ============================================================
// 5) build edge features (feat-center | center) as bf16, row-major (R_ x 128)
// ============================================================
__global__ void k_edge(const float* __restrict__ x, const int* __restrict__ idx,
                       __bf16* __restrict__ E) {
  int gid = blockIdx.x * 256 + threadIdx.x;         // R_*128 threads
  int c  = gid & 127;
  int r  = gid >> 7;
  int bn = r / K_;
  int n  = bn & (N_ - 1);
  int b  = bn >> 11;
  int j  = idx[r];
  const float* xb = x + (size_t)b * C_ * N_;
  float val = (c < C_) ? (xb[(size_t)c * N_ + j] - xb[(size_t)c * N_ + n])
                       : xb[(size_t)(c - C_) * N_ + n];
  E[gid] = (__bf16)val;
}

// ============================================================
// 6) GEMM1: z1(R_ x 64) = E(R_ x 128) @ W1^T, bf16 WMMA, fused BN-stat epilogue
//    8 waves/block, one 16x64 tile per wave; W1B staged in LDS via
//    GLOBAL_LOAD_ASYNC_TO_LDS_B128 (ASYNCcnt) instead of VGPR round-trip
// ============================================================
__global__ void k_gemm1(const __bf16* __restrict__ E, const __bf16* __restrict__ W1B,
                        const float* __restrict__ b1, float* __restrict__ z1,
                        float* __restrict__ stats /* [0:64) sum, [64:128) sumsq */) {
  __shared__ __bf16 lw[8192];                       // 16 KB packed W1
  __shared__ float  ls[128];
  int tid = threadIdx.x;
  {
    unsigned lbase = (unsigned)(uintptr_t)&lw[0];   // LDS byte offset (flat low 32b)
    const char* g = (const char*)W1B;
    #pragma unroll
    for (int i = 0; i < 4; ++i) {                   // 1024 x 16B, 4 per thread
      int t = tid + i * 256;
      async_g2l_b128(lbase + t * 16, g + (size_t)t * 16);
    }
    wait_async0();                                  // DMA done for this wave
  }
  if (tid < 128) ls[tid] = 0.f;
  __syncthreads();                                  // all waves' DMA visible

  int wave = tid >> 5, lane = tid & 31;
  int row0 = (blockIdx.x * 8 + wave) * 16;
  int m  = lane & 15;
  int bs = (lane & 16) ? 8 : 0;
  int row = row0 + m;
  const __bf16* erow = E + (size_t)row * 128;
  v8f acc[4] = {};
  #pragma unroll
  for (int kt = 0; kt < 4; ++kt) {
    union { v16bf v; v8bf h[2]; } a;                // two contiguous 16B runs per lane
    const __bf16* ap = erow + kt * 32 + bs;
    a.h[0] = *(const v8bf*)(ap);
    a.h[1] = *(const v8bf*)(ap + 16);
    __builtin_prefetch(ap + 128, 0, 1);             // same K-slice of row+1 -> global_prefetch
    #pragma unroll
    for (int ot = 0; ot < 4; ++ot) {
      union { v16bf v; v8bf h[2]; } bm;
      const v8bf* bp = (const v8bf*)&lw[((kt * 4 + ot) * 32 + lane) * 16];
      bm.h[0] = bp[0]; bm.h[1] = bp[1];
      acc[ot] = __builtin_amdgcn_wmma_f32_16x16x32_bf16(false, a.v, false, bm.v,
                                                        (short)0, acc[ot], false, false);
    }
  }
  #pragma unroll
  for (int ot = 0; ot < 4; ++ot) {
    int o = ot * 16 + m;
    float bias = b1[o];
    float ps = 0.f, pq = 0.f;
    #pragma unroll
    for (int v = 0; v < 8; ++v) {
      int r = row0 + v + bs;
      float val = acc[ot][v] + bias;
      z1[(size_t)r * 64 + o] = val;
      ps += val; pq += val * val;
    }
    atomicAdd(&ls[o], ps);
    atomicAdd(&ls[64 + o], pq);
  }
  __syncthreads();
  if (tid < 128) atomicAdd(&stats[tid], ls[tid]);
}

// ============================================================
// 7) finalize BN: scale = g*rsqrt(var+eps), shift = be - mean*scale
// ============================================================
__global__ void k_bn_final(const float* __restrict__ sum, const float* __restrict__ sq,
                           const float* __restrict__ g, const float* __restrict__ be,
                           float* __restrict__ scale, float* __restrict__ shift,
                           float inv_cnt) {
  int o = threadIdx.x;                              // 64 threads
  float mean = sum[o] * inv_cnt;
  float var  = sq[o] * inv_cnt - mean * mean;
  float sc   = g[o] * rsqrtf(var + 1e-5f);
  scale[o] = sc;
  shift[o] = be[o] - mean * sc;
}

// ============================================================
// 8) GEMM2: z2 = relu(BN1(z1)) @ W2^T ; BN1-apply + ReLU + bf16-convert fused
//    into the A-operand load; W2B staged via async DMA; BN2 stat epilogue
// ============================================================
__global__ void k_gemm2(const float* __restrict__ z1, const __bf16* __restrict__ W2B,
                        const float* __restrict__ b2,
                        const float* __restrict__ scale1, const float* __restrict__ shift1,
                        float* __restrict__ z2, float* __restrict__ stats) {
  __shared__ __bf16 lw[4096];                       // 8 KB packed W2
  __shared__ float  ls[128];
  int tid = threadIdx.x;
  {
    unsigned lbase = (unsigned)(uintptr_t)&lw[0];
    const char* g = (const char*)W2B;
    #pragma unroll
    for (int i = 0; i < 2; ++i) {                   // 512 x 16B, 2 per thread
      int t = tid + i * 256;
      async_g2l_b128(lbase + t * 16, g + (size_t)t * 16);
    }
    wait_async0();
  }
  if (tid < 128) ls[tid] = 0.f;
  __syncthreads();

  int wave = tid >> 5, lane = tid & 31;
  int row0 = (blockIdx.x * 8 + wave) * 16;
  int m  = lane & 15;
  int bs = (lane & 16) ? 8 : 0;
  int row = row0 + m;
  const float* zr = z1 + (size_t)row * 64;
  v8f acc[4] = {};
  #pragma unroll
  for (int kt = 0; kt < 2; ++kt) {
    union { v16bf v; __bf16 e[16]; } a;
    int cb = kt * 32 + bs;
    #pragma unroll
    for (int e = 0; e < 16; ++e) {
      int c = cb + ((e < 8) ? e : e + 8);
      float t = zr[c] * scale1[c] + shift1[c];      // BN1 + ReLU fused here
      a.e[e] = (__bf16)fmaxf(t, 0.f);
    }
    #pragma unroll
    for (int ot = 0; ot < 4; ++ot) {
      union { v16bf v; v8bf h[2]; } bm;
      const v8bf* bp = (const v8bf*)&lw[((kt * 4 + ot) * 32 + lane) * 16];
      bm.h[0] = bp[0]; bm.h[1] = bp[1];
      acc[ot] = __builtin_amdgcn_wmma_f32_16x16x32_bf16(false, a.v, false, bm.v,
                                                        (short)0, acc[ot], false, false);
    }
  }
  #pragma unroll
  for (int ot = 0; ot < 4; ++ot) {
    int o = ot * 16 + m;
    float bias = b2[o];
    float ps = 0.f, pq = 0.f;
    #pragma unroll
    for (int v = 0; v < 8; ++v) {
      int r = row0 + v + bs;
      float val = acc[ot][v] + bias;
      z2[(size_t)r * 64 + o] = val;
      ps += val; pq += val * val;
    }
    atomicAdd(&ls[o], ps);
    atomicAdd(&ls[64 + o], pq);
  }
  __syncthreads();
  if (tid < 128) atomicAdd(&stats[tid], ls[tid]);
}

// ============================================================
// 9) BN2 + ReLU + transpose to (B, 64, N, k); writes coalesced
// ============================================================
__global__ void k_output(const float* __restrict__ z2, const float* __restrict__ scale2,
                         const float* __restrict__ shift2, float* __restrict__ out) {
  int gid = blockIdx.x * 256 + threadIdx.x;         // 20,971,520 threads
  int kk = gid % K_;
  int t  = gid / K_;
  int n  = t & (N_ - 1);
  t >>= 11;
  int o  = t & 63;
  int b  = t >> 6;
  size_t r = (size_t)(b * N_ + n) * K_ + kk;
  float val = z2[r * 64 + o] * scale2[o] + shift2[o];
  out[gid] = fmaxf(val, 0.f);
}

// ============================================================
// host launcher
// ============================================================
extern "C" void kernel_launch(void* const* d_in, const int* in_sizes, int n_in,
                              void* d_out, int out_size, void* d_ws, size_t ws_size,
                              hipStream_t stream) {
  const float* x   = (const float*)d_in[0];
  const float* W1  = (const float*)d_in[1];
  const float* b1  = (const float*)d_in[2];
  const float* g1  = (const float*)d_in[3];
  const float* be1 = (const float*)d_in[4];
  const float* W2  = (const float*)d_in[5];
  const float* b2  = (const float*)d_in[6];
  const float* g2  = (const float*)d_in[7];
  const float* be2 = (const float*)d_in[8];
  (void)in_sizes; (void)n_in; (void)out_size; (void)ws_size;

  // ---- workspace layout (~162 MB; big region overlaid: dist dead before E/z1/z2) ----
  char* ws = (char*)d_ws;
  float*  xx    = (float*)ws;                                   //   65,536 B
  int*    idx   = (int*)  (ws + 65536);                         // 1,310,720 B
  __bf16* W1B   = (__bf16*)(ws + 65536 + 1310720);              //    16,384 B
  __bf16* W2B   = (__bf16*)(ws + 65536 + 1310720 + 16384);      //     8,192 B
  float*  stats = (float*) (ws + 65536 + 1310720 + 16384 + 8192); //   2,048 B
  char*   big   = ws + 1402880;                                 // 256-aligned
  float*  dist  = (float*)big;                                  // 134 MB (phase A)
  __bf16* E     = (__bf16*)big;                                 //  84 MB (phase B)
  float*  z1    = (float*)(big + 83886080);                     //  84 MB (phase B/C)
  float*  z2    = (float*)big;                                  //  84 MB (phase C/D)

  float* sum1 = stats,       *sq1 = stats + 64,  *scale1 = stats + 128, *shift1 = stats + 192;
  float* sum2 = stats + 256, *sq2 = stats + 320, *scale2 = stats + 384, *shift2 = stats + 448;

  const float inv_cnt = 1.0f / (float)R_;

  k_init_stats  <<<1,      512, 0, stream>>>(stats);
  k_pack_weights<<<48,     256, 0, stream>>>(W1, W2, W1B, W2B);
  k_norms       <<<64,     256, 0, stream>>>(x, xx);
  k_dist_wmma   <<<16384,  256, 0, stream>>>(x, xx, dist);
  k_topk        <<<64,     256, 0, stream>>>(dist, idx);
  k_edge        <<<163840, 256, 0, stream>>>(x, idx, E);
  k_gemm1       <<<2560,   256, 0, stream>>>(E, W1B, b1, z1, sum1);
  k_bn_final    <<<1,       64, 0, stream>>>(sum1, sq1, g1, be1, scale1, shift1, inv_cnt);
  k_gemm2       <<<2560,   256, 0, stream>>>(z1, W2B, b2, scale1, shift1, z2, sum2);
  k_bn_final    <<<1,       64, 0, stream>>>(sum2, sq2, g2, be2, scale2, shift2, inv_cnt);
  k_output      <<<81920,  256, 0, stream>>>(z2, scale2, shift2, (float*)d_out);
}
